// SingleOnlineBestP_41884521071188
// MI455X (gfx1250) — matrix operand order repaired and checked by reading, not compiled, not run on hardware
//
#include <hip/hip_runtime.h>
#include <hip/hip_bf16.h>

typedef __attribute__((ext_vector_type(16))) __bf16 v16bf;
typedef __attribute__((ext_vector_type(8)))  __bf16 v8bf;
typedef __attribute__((ext_vector_type(8)))  float  v8f;

#define B_ 16
#define S_ 512
#define C_ 16
#define D_ 512

union BFrag { v16bf v; v8bf h[2]; };

// ---------------- normalize vis: (B,S,D) f32 -> bf16, wave per vector ----------------
__global__ __launch_bounds__(256) void norm_vis_k(const float* __restrict__ vis,
                                                  __bf16* __restrict__ vnB) {
  const int wave = threadIdx.x >> 5, lane = threadIdx.x & 31;
  const size_t vec = (size_t)blockIdx.x * 8 + wave;   // < B*S
  const float* src = vis + vec * D_;
  __bf16* dst = vnB + vec * D_;
  float x[16]; float ss = 0.f;
#pragma unroll
  for (int i = 0; i < 16; ++i) { x[i] = src[i * 32 + lane]; ss += x[i] * x[i]; }
#pragma unroll
  for (int m = 1; m < 32; m <<= 1) ss += __shfl_xor(ss, m, 32);
  const float sc = 1.0f / fmaxf(sqrtf(ss), 1e-8f);
#pragma unroll
  for (int i = 0; i < 16; ++i) dst[i * 32 + lane] = (__bf16)(x[i] * sc);
}

// ---- normalize + transpose text: (B,S,C,D) f32 -> (B,C,S,D) bf16, wave per vector ----
__global__ __launch_bounds__(256) void norm_text_k(const float* __restrict__ text,
                                                   __bf16* __restrict__ tnB) {
  const int wave = threadIdx.x >> 5, lane = threadIdx.x & 31;
  const size_t vec = (size_t)blockIdx.x * 8 + wave;   // b*S*C + t*C + p
  const int b = (int)(vec >> 13);          // S*C = 8192
  const int rem = (int)(vec & 8191);
  const int t = rem >> 4;                  // C = 16
  const int p = rem & 15;
  const float* src = text + vec * D_;
  __bf16* dst = tnB + ((((size_t)b * C_ + p) * S_) + t) * D_;
  float x[16]; float ss = 0.f;
#pragma unroll
  for (int i = 0; i < 16; ++i) { x[i] = src[i * 32 + lane]; ss += x[i] * x[i]; }
#pragma unroll
  for (int m = 1; m < 32; m <<= 1) ss += __shfl_xor(ss, m, 32);
  const float sc = 1.0f / fmaxf(sqrtf(ss), 1e-8f);
#pragma unroll
  for (int i = 0; i < 16; ++i) dst[i * 32 + lane] = (__bf16)(x[i] * sc);
}

// ---------------- per-(b,p) 512x512x512 WMMA GEMM with streamed LSE ----------------
// 8 waves in 2(M) x 4(N) grid; each wave does a 64x64 register tile (16 accumulators).
// WG macro tile: 128(M) x 256(N); loops: 4 vblk x 2 tblk.
__global__ __launch_bounds__(256) void gemm_lse_k(const __bf16* __restrict__ vnB,
                                                  const __bf16* __restrict__ tnB,
                                                  float* __restrict__ clip) {
  __shared__ float rowsum[S_];
  __shared__ float colsum[S_];
  __shared__ float diagS[S_];
  __shared__ float wred[8];

  const int tid = threadIdx.x;
  const int wave = tid >> 5, lane = tid & 31;
  const int lanelo = lane & 15, laneh = lane >> 4;
  const int mw = wave & 1, nw = wave >> 1;
  const int bp = blockIdx.x;               // b*C + p
  const int b  = bp >> 4;
  const __bf16* Ab = vnB + (size_t)b  * S_ * D_;
  const __bf16* Bb = tnB + (size_t)bp * S_ * D_;

  for (int i = tid; i < S_; i += 256) { rowsum[i] = 0.f; colsum[i] = 0.f; }
  __syncthreads();

  for (int vblk = 0; vblk < 4; ++vblk) {
    const int rowBase = vblk * 128 + mw * 64;
    // A 16x32 bf16 fragment: lane row = rowBase+mt*16+lanelo; k-chunks at 8*laneh and +16
    const __bf16* ap0 = Ab + (size_t)(rowBase + lanelo) * D_ + laneh * 8;

    for (int tblk = 0; tblk < 2; ++tblk) {
      const int colBase = tblk * 256 + nw * 64;
      // B 32x16 bf16 fragment: lane col = colBase+nt*16+lanelo; k-chunks at 8*laneh and +16
      const __bf16* bp0 = Bb + (size_t)(colBase + lanelo) * D_ + laneh * 8;

      v8f acc[4][4];
#pragma unroll
      for (int mt = 0; mt < 4; ++mt)
#pragma unroll
        for (int nt = 0; nt < 4; ++nt) acc[mt][nt] = (v8f){};

#pragma unroll 2
      for (int kb = 0; kb < D_; kb += 32) {
        BFrag fa[4], fb[4];
#pragma unroll
        for (int mt = 0; mt < 4; ++mt) {
          const __bf16* p = ap0 + (size_t)(mt * 16) * D_ + kb;
          fa[mt].h[0] = *(const v8bf*)(p);
          fa[mt].h[1] = *(const v8bf*)(p + 16);
        }
#pragma unroll
        for (int nt = 0; nt < 4; ++nt) {
          const __bf16* q = bp0 + (size_t)(nt * 16) * D_ + kb;
          fb[nt].h[0] = *(const v8bf*)(q);
          fb[nt].h[1] = *(const v8bf*)(q + 16);
        }
#pragma unroll
        for (int mt = 0; mt < 4; ++mt)
#pragma unroll
          for (int nt = 0; nt < 4; ++nt)
            acc[mt][nt] = __builtin_amdgcn_wmma_f32_16x16x32_bf16(
                false, fa[mt].v, false, fb[nt].v, (short)0, acc[mt][nt], false, false);
      }

      // epilogue: exp in-register; per-mt row partials + butterfly, cols + diag to LDS
      float colp[4] = {0.f, 0.f, 0.f, 0.f};
#pragma unroll
      for (int mt = 0; mt < 4; ++mt) {
        const int rb = rowBase + mt * 16 + laneh * 8;
        float rowp8[8];
#pragma unroll
        for (int i = 0; i < 8; ++i) rowp8[i] = 0.f;
#pragma unroll
        for (int nt = 0; nt < 4; ++nt) {
          const int cb = colBase + nt * 16 + lanelo;
          float cs = 0.f;
#pragma unroll
          for (int i = 0; i < 8; ++i) {
            const float s = acc[mt][nt][i];
            if (rb + i == cb) diagS[cb] = s;
            const float e = __expf(s);
            rowp8[i] += e;
            cs += e;
          }
          colp[nt] += cs;
        }
        // flush row partials: 16-lane butterfly, lanes 0/16 commit
#pragma unroll
        for (int i = 0; i < 8; ++i) {
          float v = rowp8[i];
          v += __shfl_xor(v, 1, 32);
          v += __shfl_xor(v, 2, 32);
          v += __shfl_xor(v, 4, 32);
          v += __shfl_xor(v, 8, 32);
          if (lanelo == 0) atomicAdd(&rowsum[rb + i], v);
        }
      }
#pragma unroll
      for (int nt = 0; nt < 4; ++nt)
        atomicAdd(&colsum[colBase + nt * 16 + lanelo], colp[nt]);
    }
  }
  __syncthreads();

  // clip_loss[b,p] = mean_i( 0.5*(log rowsum + log colsum) - diag )
  float part = 0.f;
  for (int i = tid; i < S_; i += 256)
    part += 0.5f * (__logf(rowsum[i]) + __logf(colsum[i])) - diagS[i];
#pragma unroll
  for (int m = 1; m < 32; m <<= 1) part += __shfl_xor(part, m, 32);
  if (lane == 0) wred[wave] = part;
  __syncthreads();
  if (tid == 0) {
    float s = 0.f;
#pragma unroll
    for (int w = 0; w < 8; ++w) s += wred[w];
    clip[bp] = s * (1.0f / S_);
  }
}

// ---------------- loss / acc / preds ----------------
__global__ __launch_bounds__(256) void finalize_k(const float* __restrict__ output,
                                                  const float* __restrict__ clip,
                                                  float* __restrict__ out) {
  __shared__ float cl[256];
  __shared__ float red[8];
  __shared__ float matchS[16];
  __shared__ int   predS[16];
  const int tid = threadIdx.x, lane = tid & 31, wave = tid >> 5;
  const float o = output[tid];
  const float c = clip[tid];
  cl[tid] = c;
  float v = (o - c) * (o - c);
#pragma unroll
  for (int m = 1; m < 32; m <<= 1) v += __shfl_xor(v, m, 32);
  if (lane == 0) red[wave] = v;
  __syncthreads();
  if (tid < 16) {
    const int b = tid;
    float bo = output[b * 16]; int po = 0;
    float bc = cl[b * 16];     int pc = 0;
    for (int p = 1; p < 16; ++p) {
      const float vo = output[b * 16 + p]; if (vo < bo) { bo = vo; po = p; }
      const float vc = cl[b * 16 + p];     if (vc < bc) { bc = vc; pc = p; }
    }
    predS[b]  = po;
    matchS[b] = (po == pc) ? 1.f : 0.f;
  }
  __syncthreads();
  if (tid == 0) {
    float s = 0.f;
#pragma unroll
    for (int w = 0; w < 8; ++w) s += red[w];
    out[0] = s * (1.f / 256.f);
    float a = 0.f;
#pragma unroll
    for (int b = 0; b < 16; ++b) a += matchS[b];
    out[1] = a * (100.f / 16.f);
  }
  for (int idx = tid; idx < B_ * S_; idx += 256)
    out[2 + idx] = (float)predS[idx >> 9];
}

extern "C" void kernel_launch(void* const* d_in, const int* in_sizes, int n_in,
                              void* d_out, int out_size, void* d_ws, size_t ws_size,
                              hipStream_t stream) {
  const float* output = (const float*)d_in[0];               // (B, C)
  const float* text   = (const float*)d_in[1];               // (B, S, C, D)
  const float* vis    = (const float*)d_in[2];               // (B, S, D)
  float* out = (float*)d_out;

  char* ws = (char*)d_ws;
  const size_t tnBytes = (size_t)B_ * C_ * S_ * D_ * sizeof(__bf16);  // 128 MB
  const size_t vnBytes = (size_t)B_ * S_ * D_ * sizeof(__bf16);       // 8 MB
  __bf16* tnB  = (__bf16*)ws;
  __bf16* vnB  = (__bf16*)(ws + tnBytes);
  float*  clip = (float*)(ws + tnBytes + vnBytes);                    // 256 f32

  norm_vis_k <<<(B_ * S_) / 8,      256, 0, stream>>>(vis,  vnB);
  norm_text_k<<<(B_ * S_ * C_) / 8, 256, 0, stream>>>(text, tnB);
  gemm_lse_k <<<B_ * C_,            256, 0, stream>>>(vnB, tnB, clip);
  finalize_k <<<1,                  256, 0, stream>>>(output, clip, out);
}
